// ConditionalLSTMVAE_57543971832700
// MI455X (gfx1250) — matrix operand order repaired and checked by reading, not compiled, not run on hardware
//
#include <hip/hip_runtime.h>

// ---------------------------------------------------------------------------
// Conditional LSTM-VAE forward for MI455X (gfx1250, wave32, WMMA bf16)
//  - weights pre-swizzled to per-lane WMMA B-fragment layout (b128 loads)
//  - recurrent h state kept in bf16 (no cvt in hot loop)
//  - 2 M-tiles per wave sharing one B fragment
// ---------------------------------------------------------------------------

typedef __attribute__((ext_vector_type(16))) __bf16 v16bf;
typedef __attribute__((ext_vector_type(8)))  __bf16 v8bf;
typedef __attribute__((ext_vector_type(8)))  float  v8f;

#define B_  512
#define T_  360
#define D_  12
#define H_  256
#define L_  64
#define E_  8
#define G4_ 1024   // 4*H
#define FRAG_STRIDE (64 * 32 * 16)   // elements per k-block of a swizzled weight

// ------------------------- helpers -----------------------------------------
static __device__ __forceinline__ __bf16 f2bf(float x) {
    unsigned u = __builtin_bit_cast(unsigned, x);
    unsigned r = u + 0x7FFFu + ((u >> 16) & 1u);       // round-to-nearest-even
    unsigned short s = (unsigned short)(r >> 16);
    return __builtin_bit_cast(__bf16, s);
}
static __device__ __forceinline__ float bf2f(__bf16 b) {
    unsigned short s = __builtin_bit_cast(unsigned short, b);
    unsigned u = ((unsigned)s) << 16;
    return __builtin_bit_cast(float, u);
}
static __device__ __forceinline__ float sigmf(float x) {
    return 1.0f / (1.0f + __expf(-x));
}
// A-fragment for wmma bf16 16x16x32: 16 contiguous k per lane (k = 16*half + i)
static __device__ __forceinline__ v16bf load_a(const __bf16* base, int half) {
    v8bf lo = *(const v8bf*)(base + half * 16);
    v8bf hi = *(const v8bf*)(base + half * 16 + 8);
    return __builtin_shufflevector(lo, hi, 0, 1, 2, 3, 4, 5, 6, 7,
                                           8, 9, 10, 11, 12, 13, 14, 15);
}

// --- weight swizzle: fp32 [1024,K] row-major -> bf16 B-fragment layout ------
// WF[((kb6*64 + tn)*32 + lane)*16 + i] = W[n,k], n = tn*16+(lane&15),
//                                        k = kb6*32 + (lane>>4)*16 + i
__global__ void conv_frag_kernel(const float* __restrict__ W, __bf16* __restrict__ WF,
                                 int K, int Kpad) {
    int idx = blockIdx.x * blockDim.x + threadIdx.x;
    if (idx >= Kpad * G4_) return;
    int i    = idx & 15;
    int lane = (idx >> 4) & 31;
    int tn   = (idx >> 9) & 63;
    int kb6  = idx >> 15;
    int n = tn * 16 + (lane & 15);
    int k = kb6 * 32 + (lane >> 4) * 16 + i;
    WF[idx] = (k < K) ? f2bf(W[(size_t)n * K + k]) : f2bf(0.0f);
}

__global__ void addb_kernel(const float* a, const float* b, float* o, int n) {
    int i = blockIdx.x * blockDim.x + threadIdx.x;
    if (i < n) o[i] = a[i] + b[i];
}

__global__ void f32tobf16_kernel(const float* __restrict__ s, __bf16* __restrict__ d,
                                 int n) {
    int i = blockIdx.x * blockDim.x + threadIdx.x;
    if (i < n) d[i] = f2bf(s[i]);
}

// ------------------- input staging -----------------------------------------
__global__ void build_xc_kernel(const float* __restrict__ x, const float* __restrict__ emb,
                                const int* __restrict__ labels, __bf16* __restrict__ xc) {
    int idx = blockIdx.x * blockDim.x + threadIdx.x;
    if (idx >= B_ * T_ * 32) return;
    int j = idx & 31, row = idx >> 5, b = row / T_;
    float v = 0.0f;
    if (j < 12)       v = x[(size_t)row * D_ + j];
    else if (j < 20)  v = emb[labels[b] * E_ + (j - 12)];
    xc[idx] = f2bf(v);
}

__global__ void build_din_kernel(const float* __restrict__ x, const float* __restrict__ zc,
                                 __bf16* __restrict__ din) {
    int idx = blockIdx.x * blockDim.x + threadIdx.x;
    if (idx >= B_ * T_ * 96) return;
    int j = idx % 96, row = idx / 96, b = row / T_, t = row % T_;
    float v = 0.0f;
    if (j < 12)       v = (t == 0) ? 0.0f : x[(size_t)(row - 1) * D_ + j];
    else if (j < 84)  v = zc[b * 72 + (j - 12)];
    din[idx] = f2bf(v);
}

// ------------------- LSTM per-step gate GEMM (WMMA bf16) --------------------
// gates[B,1024] = h@WhhT + xseq[:,t,:]@WihT + bias ; h is bf16 [B,256]
// Each wave: two 16x16 M-tiles sharing one B fragment. 1024 waves, grid 128.
__global__ void __launch_bounds__(256)
lstm_gates_kernel(const __bf16* __restrict__ h, const __bf16* __restrict__ xseq,
                  int t, int xdim,
                  const __bf16* __restrict__ WhhF, const __bf16* __restrict__ WihF,
                  const float* __restrict__ bias, float* __restrict__ gates) {
    const int lane = threadIdx.x & 31;
    const int tile = blockIdx.x * 8 + (threadIdx.x >> 5);
    const int tmg  = tile >> 6;         // 0..15 -> 32-row group
    const int tn   = tile & 63;         // 0..63 -> 16-col tile
    const int half = lane >> 4;
    const int l15  = lane & 15;
    const int m0 = tmg * 32 + l15;
    const int m1 = m0 + 16;
    const int n  = tn * 16 + l15;

    v8f acc0 = {}, acc1 = {};

    const __bf16* h0 = h + (size_t)m0 * H_;
    const __bf16* h1 = h + (size_t)m1 * H_;
    const __bf16* wf = WhhF + ((size_t)tn * 32 + lane) * 16;
#pragma unroll
    for (int kb6 = 0; kb6 < H_ / 32; ++kb6) {
        v16bf b  = *(const v16bf*)(wf + (size_t)kb6 * FRAG_STRIDE);
        v16bf a0 = load_a(h0 + kb6 * 32, half);
        v16bf a1 = load_a(h1 + kb6 * 32, half);
        acc0 = __builtin_amdgcn_wmma_f32_16x16x32_bf16(false, a0, false, b,
                                                       (short)0, acc0, false, false);
        acc1 = __builtin_amdgcn_wmma_f32_16x16x32_bf16(false, a1, false, b,
                                                       (short)0, acc1, false, false);
    }
    const __bf16* x0 = xseq + ((size_t)m0 * T_ + t) * xdim;
    const __bf16* x1 = xseq + ((size_t)m1 * T_ + t) * xdim;
    const __bf16* wx = WihF + ((size_t)tn * 32 + lane) * 16;
    for (int kb6 = 0; kb6 < xdim / 32; ++kb6) {
        v16bf b  = *(const v16bf*)(wx + (size_t)kb6 * FRAG_STRIDE);
        v16bf a0 = load_a(x0 + kb6 * 32, half);
        v16bf a1 = load_a(x1 + kb6 * 32, half);
        acc0 = __builtin_amdgcn_wmma_f32_16x16x32_bf16(false, a0, false, b,
                                                       (short)0, acc0, false, false);
        acc1 = __builtin_amdgcn_wmma_f32_16x16x32_bf16(false, a1, false, b,
                                                       (short)0, acc1, false, false);
    }
    const float bn = bias[n];
#pragma unroll
    for (int r = 0; r < 8; ++r) {
        gates[(size_t)(tmg * 32 + r + 8 * half) * G4_ + n]      = acc0[r] + bn;
        gates[(size_t)(tmg * 32 + 16 + r + 8 * half) * G4_ + n] = acc1[r] + bn;
    }
}

// ------------------- LSTM pointwise (i,f,g,o) -------------------------------
__global__ void lstm_point_kernel(const float* __restrict__ gates,
                                  float* __restrict__ c, __bf16* __restrict__ h,
                                  __bf16* __restrict__ outseq, int t,
                                  int ostride, int ooff) {
    int idx = blockIdx.x * blockDim.x + threadIdx.x;
    if (idx >= B_ * H_) return;
    int b = idx >> 8, j = idx & 255;
    const float* g = gates + (size_t)b * G4_;
    float ig = sigmf(g[j]);
    float fg = sigmf(g[H_ + j]);
    float gg = tanhf(g[2 * H_ + j]);
    float og = sigmf(g[3 * H_ + j]);
    float cn = fg * c[idx] + ig * gg;
    float hn = og * tanhf(cn);
    c[idx] = cn;
    __bf16 hb = f2bf(hn);
    h[idx] = hb;
    if (outseq)
        outseq[((size_t)b * T_ + t) * ostride + ooff + j] = hb;
}

// ------------------- small dense layers ------------------------------------
__global__ void dense_kernel(const float* __restrict__ in, const float* __restrict__ W,
                             const float* __restrict__ bias, float* __restrict__ out,
                             int K, int L) {
    int idx = blockIdx.x * blockDim.x + threadIdx.x;
    if (idx >= B_ * L) return;
    int b = idx / L, l = idx % L;
    float acc = bias[l];
    const float* r = in + (size_t)b * K;
    const float* w = W + (size_t)l * K;
    for (int k = 0; k < K; ++k) acc += r[k] * w[k];
    out[idx] = acc;
}

__global__ void z_zc_kernel(const float* __restrict__ eps, const float* __restrict__ emb,
                            const int* __restrict__ labels,
                            const float* __restrict__ mu, const float* __restrict__ lv,
                            float* __restrict__ zout, float* __restrict__ zc) {
    int idx = blockIdx.x * blockDim.x + threadIdx.x;
    if (idx >= B_ * 72) return;
    int b = idx / 72, j = idx % 72;
    if (j < L_) {
        float m = mu[b * L_ + j], v = lv[b * L_ + j];
        float z = m + eps[b * L_ + j] * __expf(0.5f * v);
        zout[b * L_ + j] = z;
        zc[b * 72 + j] = z;
    } else {
        zc[b * 72 + j] = emb[labels[b] * E_ + (j - L_)];
    }
}

__global__ void copy_cols_kernel(const __bf16* __restrict__ h, float* __restrict__ hcat,
                                 int off) {
    int idx = blockIdx.x * blockDim.x + threadIdx.x;
    if (idx >= B_ * H_) return;
    int b = idx >> 8, j = idx & 255;
    hcat[(size_t)b * 512 + off + j] = bf2f(h[idx]);
}

// ------------------- final projection: y1d[B,T,256] @ Wout^T + bout ---------
__global__ void proj_kernel(const __bf16* __restrict__ y1d, const float* __restrict__ Wout,
                            const float* __restrict__ bout, float* __restrict__ out) {
    int idx = blockIdx.x * blockDim.x + threadIdx.x;
    if (idx >= B_ * T_ * D_) return;
    int d = idx % D_;
    size_t row = idx / D_;
    float acc = bout[d];
    const __bf16* r = y1d + row * H_;
    const float* w = Wout + (size_t)d * H_;
    for (int k = 0; k < H_; ++k) acc += bf2f(r[k]) * w[k];
    out[idx] = acc;
}

// ---------------------------------------------------------------------------
extern "C" void kernel_launch(void* const* d_in, const int* in_sizes, int n_in,
                              void* d_out, int out_size, void* d_ws, size_t ws_size,
                              hipStream_t stream) {
    const float* x      = (const float*)d_in[0];
    const int*   labels = (const int*)d_in[1];
    const float* eps    = (const float*)d_in[2];
    const float* emb    = (const float*)d_in[3];
    const float* Wih0f = (const float*)d_in[4],  *Whh0f = (const float*)d_in[5];
    const float* bih0f = (const float*)d_in[6],  *bhh0f = (const float*)d_in[7];
    const float* Wih0b = (const float*)d_in[8],  *Whh0b = (const float*)d_in[9];
    const float* bih0b = (const float*)d_in[10], *bhh0b = (const float*)d_in[11];
    const float* Wih1f = (const float*)d_in[12], *Whh1f = (const float*)d_in[13];
    const float* bih1f = (const float*)d_in[14], *bhh1f = (const float*)d_in[15];
    const float* Wih1b = (const float*)d_in[16], *Whh1b = (const float*)d_in[17];
    const float* bih1b = (const float*)d_in[18], *bhh1b = (const float*)d_in[19];
    const float* Wmu  = (const float*)d_in[20], *bmu  = (const float*)d_in[21];
    const float* Wlv  = (const float*)d_in[22], *blv  = (const float*)d_in[23];
    const float* Whid = (const float*)d_in[24], *bhid = (const float*)d_in[25];
    const float* Wcel = (const float*)d_in[26], *bcel = (const float*)d_in[27];
    const float* dWih0 = (const float*)d_in[28], *dWhh0 = (const float*)d_in[29];
    const float* dbih0 = (const float*)d_in[30], *dbhh0 = (const float*)d_in[31];
    const float* dWih1 = (const float*)d_in[32], *dWhh1 = (const float*)d_in[33];
    const float* dbih1 = (const float*)d_in[34], *dbhh1 = (const float*)d_in[35];
    const float* Wout = (const float*)d_in[36], *bout = (const float*)d_in[37];

    float* dout = (float*)d_out;
    const size_t mu_off = (size_t)B_ * T_ * D_;
    const size_t lv_off = mu_off + (size_t)B_ * L_;
    const size_t z_off  = lv_off + (size_t)B_ * L_;

    // -------- workspace carve-out (deterministic) --------
    char* w = (char*)d_ws;
    size_t cur = 0;
    auto A = [&](size_t bytes) -> void* {
        void* p = w + cur;
        cur = (cur + bytes + 255) & ~(size_t)255;
        return p;
    };
    __bf16* whh0f_t = (__bf16*)A((size_t)256 * G4_ * 2);
    __bf16* whh0b_t = (__bf16*)A((size_t)256 * G4_ * 2);
    __bf16* whh1f_t = (__bf16*)A((size_t)256 * G4_ * 2);
    __bf16* whh1b_t = (__bf16*)A((size_t)256 * G4_ * 2);
    __bf16* dwhh0_t = (__bf16*)A((size_t)256 * G4_ * 2);
    __bf16* dwhh1_t = (__bf16*)A((size_t)256 * G4_ * 2);
    __bf16* wih0f_t = (__bf16*)A((size_t)32  * G4_ * 2);
    __bf16* wih0b_t = (__bf16*)A((size_t)32  * G4_ * 2);
    __bf16* wih1f_t = (__bf16*)A((size_t)512 * G4_ * 2);
    __bf16* wih1b_t = (__bf16*)A((size_t)512 * G4_ * 2);
    __bf16* dwih0_t = (__bf16*)A((size_t)96  * G4_ * 2);
    __bf16* dwih1_t = (__bf16*)A((size_t)256 * G4_ * 2);
    float* bias0f = (float*)A(G4_ * 4);
    float* bias0b = (float*)A(G4_ * 4);
    float* bias1f = (float*)A(G4_ * 4);
    float* bias1b = (float*)A(G4_ * 4);
    float* dbias0 = (float*)A(G4_ * 4);
    float* dbias1 = (float*)A(G4_ * 4);
    __bf16* xc  = (__bf16*)A((size_t)B_ * T_ * 32  * 2);
    __bf16* y0  = (__bf16*)A((size_t)B_ * T_ * 512 * 2);
    __bf16* din = (__bf16*)A((size_t)B_ * T_ * 96  * 2);
    __bf16* y0d = (__bf16*)A((size_t)B_ * T_ * 256 * 2);
    __bf16* y1d = (__bf16*)A((size_t)B_ * T_ * 256 * 2);
    float* gatesA = (float*)A((size_t)B_ * G4_ * 4);
    float* gatesB = (float*)A((size_t)B_ * G4_ * 4);
    __bf16* hF = (__bf16*)A((size_t)B_ * H_ * 2);
    __bf16* hB = (__bf16*)A((size_t)B_ * H_ * 2);
    __bf16* hd0 = (__bf16*)A((size_t)B_ * H_ * 2);
    __bf16* hd1 = (__bf16*)A((size_t)B_ * H_ * 2);
    float* cF = (float*)A((size_t)B_ * H_ * 4);
    float* cB = (float*)A((size_t)B_ * H_ * 4);
    float* hcat = (float*)A((size_t)B_ * 512 * 4);
    float* zc   = (float*)A((size_t)B_ * 72 * 4);
    float* hidb = (float*)A((size_t)B_ * 512 * 4);   // [B,512] == reshape(2,B,256)
    float* celb = (float*)A((size_t)B_ * 512 * 4);

    // -------- weight swizzle to WMMA B-fragment layout --------
    auto conv = [&](const float* W, __bf16* WF, int K, int Kpad) {
        int total = Kpad * G4_;
        conv_frag_kernel<<<(total + 255) / 256, 256, 0, stream>>>(W, WF, K, Kpad);
    };
    conv(Whh0f, whh0f_t, 256, 256);  conv(Whh0b, whh0b_t, 256, 256);
    conv(Whh1f, whh1f_t, 256, 256);  conv(Whh1b, whh1b_t, 256, 256);
    conv(dWhh0, dwhh0_t, 256, 256);  conv(dWhh1, dwhh1_t, 256, 256);
    conv(Wih0f, wih0f_t, 20, 32);    conv(Wih0b, wih0b_t, 20, 32);
    conv(Wih1f, wih1f_t, 512, 512);  conv(Wih1b, wih1b_t, 512, 512);
    conv(dWih0, dwih0_t, 84, 96);    conv(dWih1, dwih1_t, 256, 256);
    addb_kernel<<<4, 256, 0, stream>>>(bih0f, bhh0f, bias0f, G4_);
    addb_kernel<<<4, 256, 0, stream>>>(bih0b, bhh0b, bias0b, G4_);
    addb_kernel<<<4, 256, 0, stream>>>(bih1f, bhh1f, bias1f, G4_);
    addb_kernel<<<4, 256, 0, stream>>>(bih1b, bhh1b, bias1b, G4_);
    addb_kernel<<<4, 256, 0, stream>>>(dbih0, dbhh0, dbias0, G4_);
    addb_kernel<<<4, 256, 0, stream>>>(dbih1, dbhh1, dbias1, G4_);

    // -------- stage encoder input --------
    {
        int total = B_ * T_ * 32;
        build_xc_kernel<<<(total + 255) / 256, 256, 0, stream>>>(x, emb, labels, xc);
    }

    // bidirectional LSTM pass (fwd + bwd per step)
    auto run_bidir = [&](const __bf16* xseq, int xdim,
                         const __bf16* WhhTf, const __bf16* WihTf, const float* bf,
                         const __bf16* WhhTb, const __bf16* WihTb, const float* bb,
                         __bf16* outseq, int ostride) {
        hipMemsetAsync(hF, 0, (size_t)B_ * H_ * 2, stream);
        hipMemsetAsync(hB, 0, (size_t)B_ * H_ * 2, stream);
        hipMemsetAsync(cF, 0, (size_t)B_ * H_ * 4, stream);
        hipMemsetAsync(cB, 0, (size_t)B_ * H_ * 4, stream);
        for (int s = 0; s < T_; ++s) {
            int tf = s, tb = T_ - 1 - s;
            lstm_gates_kernel<<<128, 256, 0, stream>>>(hF, xseq, tf, xdim, WhhTf, WihTf, bf, gatesA);
            lstm_gates_kernel<<<128, 256, 0, stream>>>(hB, xseq, tb, xdim, WhhTb, WihTb, bb, gatesB);
            lstm_point_kernel<<<512, 256, 0, stream>>>(gatesA, cF, hF, outseq, tf, ostride, 0);
            lstm_point_kernel<<<512, 256, 0, stream>>>(gatesB, cB, hB, outseq, tb, ostride, H_);
        }
    };
    auto run_uni = [&](__bf16* h, float* c, const __bf16* xseq, int xdim,
                       const __bf16* WhhT, const __bf16* WihT, const float* b,
                       __bf16* outseq, int ostride) {
        for (int s = 0; s < T_; ++s) {
            lstm_gates_kernel<<<128, 256, 0, stream>>>(h, xseq, s, xdim, WhhT, WihT, b, gatesA);
            lstm_point_kernel<<<512, 256, 0, stream>>>(gatesA, c, h, outseq, s, ostride, 0);
        }
    };

    // -------- encoder layer 0 (bidir) -> y0 [B,T,512] bf16 --------
    run_bidir(xc, 32, whh0f_t, wih0f_t, bias0f, whh0b_t, wih0b_t, bias0b, y0, 512);
    // -------- encoder layer 1 (bidir, final states only) --------
    run_bidir(y0, 512, whh1f_t, wih1f_t, bias1f, whh1b_t, wih1b_t, bias1b,
              (__bf16*)nullptr, 0);
    copy_cols_kernel<<<512, 256, 0, stream>>>(hF, hcat, 0);
    copy_cols_kernel<<<512, 256, 0, stream>>>(hB, hcat, H_);

    // -------- VAE bottleneck --------
    dense_kernel<<<(B_ * L_ + 255) / 256, 256, 0, stream>>>(hcat, Wmu, bmu, dout + mu_off, 512, L_);
    dense_kernel<<<(B_ * L_ + 255) / 256, 256, 0, stream>>>(hcat, Wlv, blv, dout + lv_off, 512, L_);
    z_zc_kernel<<<(B_ * 72 + 255) / 256, 256, 0, stream>>>(eps, emb, labels,
                                                           dout + mu_off, dout + lv_off,
                                                           dout + z_off, zc);
    // hid/cel: [B,512] result == row-major reshape(2,B,256); use blocks directly
    dense_kernel<<<(B_ * 512 + 255) / 256, 256, 0, stream>>>(zc, Whid, bhid, hidb, 72, 512);
    dense_kernel<<<(B_ * 512 + 255) / 256, 256, 0, stream>>>(zc, Wcel, bcel, celb, 72, 512);
    // decoder initial h states in bf16
    f32tobf16_kernel<<<512, 256, 0, stream>>>(hidb, hd0, B_ * H_);
    f32tobf16_kernel<<<512, 256, 0, stream>>>(hidb + (size_t)B_ * H_, hd1, B_ * H_);

    // -------- decoder input staging --------
    {
        int total = B_ * T_ * 96;
        build_din_kernel<<<(total + 255) / 256, 256, 0, stream>>>(x, zc, din);
    }

    // -------- decoder layer 0 / layer 1 --------
    run_uni(hd0, celb, din, 96, dwhh0_t, dwih0_t, dbias0, y0d, H_);
    run_uni(hd1, celb + (size_t)B_ * H_, y0d, 256, dwhh1_t, dwih1_t, dbias1, y1d, H_);

    // -------- output projection --------
    {
        int total = B_ * T_ * D_;
        proj_kernel<<<(total + 255) / 256, 256, 0, stream>>>(y1d, Wout, bout, dout);
    }
}
